// TimestepVisionTransformer_75986561401154
// MI455X (gfx1250) — matrix operand order repaired and checked by reading, not compiled, not run on hardware
//
#include <hip/hip_runtime.h>
#include <cstdint>
#include <cstddef>

// ---------------- problem constants ----------------
#define BB    128
#define IMGSZ 224
#define PATCH 16
#define EE    96
#define HHEADS 4
#define DDIM  24
#define NAXIS 14
#define NNTOK 196            // 14*14
#define NT    (BB*NNTOK)     // 25088 tokens
#define KPATCH 256           // 16*16*1

typedef __attribute__((ext_vector_type(16))) _Float16 v16h;
typedef __attribute__((ext_vector_type(8)))  _Float16 v8h;
typedef __attribute__((ext_vector_type(8)))  float    v8f;

// fast silu: x * rcp(1+exp(-x))  (v_exp_f32 + v_rcp_f32, no IEEE divide expansion)
__device__ __forceinline__ float silu_f(float v)
{
    return v * __builtin_amdgcn_rcpf(1.0f + __expf(-v));
}

union V16U { v16h v; v8h h[2]; };

// ---------------- WMMA GEMM ----------------
// C[M,N] = A[M,K](f16, row-major) x B (given TRANSPOSED: Bt[N][K], f16) + bias[N].
// Optional SiLU; f32 or f16 output. One wave per block; each wave computes a
// 64(M) x 16(N) macro-tile = 4 WMMA tiles sharing one B fragment per k-step.
// K compile-time (full unroll). All fragments built from 128-bit loads.
template<int K, int ACT, bool OUT16>
__global__ __launch_bounds__(32)
void wmma_gemm_t(const _Float16* __restrict__ A, const _Float16* __restrict__ Bt,
                 const float* __restrict__ bias, float* __restrict__ Cf,
                 _Float16* __restrict__ Ch, int Nn)
{
    const int tileM = blockIdx.x * 64;
    const int tileN = blockIdx.y * 16;
    const int lane  = threadIdx.x;      // 0..31
    const int half  = lane >> 4;        // 0 | 1
    const int l16   = lane & 15;

    // per-lane base pointers (computed once; k-loop adds constant offsets)
    const _Float16* Bbase = Bt + (size_t)(tileN + l16) * K + half * 16;
    const _Float16* Abase = A + (size_t)(tileM + l16) * K + half * 16;

    v8f acc[4] = {};
    #pragma unroll
    for (int k0 = 0; k0 < K; k0 += 32) {
        // B fragment: lane col = l16, element h -> K = k0 + 16*half + h (contiguous)
        V16U b;
        b.h[0] = *(const v8h*)(Bbase + k0);
        b.h[1] = *(const v8h*)(Bbase + k0 + 8);

        #pragma unroll
        for (int m = 0; m < 4; ++m) {
            // A fragment: lane row = l16, element h -> K = k0 + 16*half + h (contiguous)
            V16U a;
            const _Float16* Arow = Abase + (size_t)(m * 16) * K + k0;
            a.h[0] = *(const v8h*)(Arow);
            a.h[1] = *(const v8h*)(Arow + 8);
            acc[m] = __builtin_amdgcn_wmma_f32_16x16x32_f16(
                         false, a.v, false, b.v, (short)0, acc[m], false, false);
        }
    }
    // D element r -> row tileM + m*16 + r + 8*half, col tileN + l16
    const int col = tileN + l16;
    const float bv = bias[col];
    #pragma unroll
    for (int m = 0; m < 4; ++m) {
        #pragma unroll
        for (int r = 0; r < 8; ++r) {
            int row = tileM + m * 16 + r + 8 * half;
            float v = acc[m][r] + bv;
            if (ACT == 1) v = silu_f(v);
            if (OUT16) Ch[(size_t)row * Nn + col] = (_Float16)v;
            else       Cf[(size_t)row * Nn + col] = v;
        }
    }
}

// ---------------- time embedding (tiny) ----------------
__global__ void time_embed_kernel(const float* __restrict__ ts,
                                  const float* __restrict__ Wt1, const float* __restrict__ bt1,
                                  const float* __restrict__ Wt2, const float* __restrict__ bt2,
                                  float* __restrict__ timeo)
{
    __shared__ float h[128];
    float e = 0.6931471805599453f * ts[0];
    float sv = sinf(e), cv = cosf(e);
    int j = threadIdx.x;
    if (j < 128) {
        float v = sv * Wt1[j] + cv * Wt1[128 + j] + bt1[j];
        h[j] = silu_f(v);
    }
    __syncthreads();
    if (j < EE) {
        float acc = bt2[j];
        for (int k = 0; k < 128; ++k) acc += h[k] * Wt2[k * EE + j];
        timeo[j] = acc;
    }
}

// ---------------- conversions ----------------
__global__ void f32_to_f16_kernel(const float* __restrict__ in, _Float16* __restrict__ out, size_t n)
{
    size_t i = (size_t)blockIdx.x * blockDim.x + threadIdx.x;
    if (i < n) out[i] = (_Float16)in[i];
}

// in[rows][cols] -> out[cols][rows], f16  (used to store weights as B^T[N][K])
__global__ void transpose_f32_to_f16_kernel(const float* __restrict__ in, _Float16* __restrict__ out,
                                            int rows, int cols)
{
    size_t i = (size_t)blockIdx.x * blockDim.x + threadIdx.x;
    if (i >= (size_t)rows * cols) return;
    int c = (int)(i % cols);
    int r = (int)(i / cols);
    out[(size_t)c * rows + r] = (_Float16)in[i];
}

// ---------------- im2col of 16x16 patches (C_IN = 1) ----------------
__global__ void im2col_patch_kernel(const float* __restrict__ x, _Float16* __restrict__ A)
{
    size_t idx = (size_t)blockIdx.x * blockDim.x + threadIdx.x;
    if (idx >= (size_t)NT * KPATCH) return;
    int k = (int)(idx & (KPATCH - 1));
    size_t t = idx >> 8;                 // token id = b*196 + n
    int b = (int)(t / NNTOK);
    int n = (int)(t % NNTOK);
    int pr = n / NAXIS, pc = n % NAXIS;
    int py = k >> 4,    px = k & 15;
    int iy = pr * PATCH + py, ix = pc * PATCH + px;
    A[idx] = (_Float16)x[((size_t)b * IMGSZ + iy) * IMGSZ + ix];
}

// ---------------- rotary position embedding (in-place on p[B,N,E]) ----------------
__global__ void rope_kernel(float* __restrict__ p)
{
    size_t idx = (size_t)blockIdx.x * blockDim.x + threadIdx.x;
    if (idx >= (size_t)NT * (EE / 2)) return;
    int i = (int)(idx % (EE / 2));
    size_t t = idx / (EE / 2);
    int n = (int)(t % NNTOK);
    int r = n / NAXIS, c = n % NAXIS;
    float theta = powf(10000.0f, -2.0f * (float)i / (float)EE);
    float ang = theta * (float)(r + c);
    float cs = cosf(ang), sn = sinf(ang);
    float* row = p + t * EE;
    float xe = row[2 * i], ye = row[2 * i + 1];
    row[2 * i]     = xe * cs - ye * sn;
    row[2 * i + 1] = xe * sn + ye * cs;
}

// ---------------- per-token LayerNorm over E, emit f16 ----------------
__global__ void ln_tokens_f16_kernel(const float* __restrict__ p, const float* __restrict__ g,
                                     const float* __restrict__ bt, _Float16* __restrict__ out)
{
    size_t t = (size_t)blockIdx.x * blockDim.x + threadIdx.x;
    if (t >= NT) return;
    const float* row = p + t * EE;
    float mu = 0.f;
    for (int e = 0; e < EE; ++e) mu += row[e];
    mu *= (1.0f / EE);
    float var = 0.f;
    for (int e = 0; e < EE; ++e) { float d = row[e] - mu; var += d * d; }
    var *= (1.0f / EE);
    float rs = rsqrtf(var + 1e-5f);
    _Float16* orow = out + t * EE;
    for (int e = 0; e < EE; ++e)
        orow[e] = (_Float16)((row[e] - mu) * rs * g[e] + bt[e]);
}

// ------ fused attention per (batch, head): K/V in LDS, streaming softmax ------
// writes head-concatenated output directly as f16 (feeds the Wo WMMA GEMM)
__global__ __launch_bounds__(256)
void attention_kernel(const float* __restrict__ qkv, _Float16* __restrict__ o16)
{
    int bh = blockIdx.x;
    int b = bh >> 2, h = bh & 3;
    __shared__ float Ks[NNTOK][DDIM];   // 18.8 KB
    __shared__ float Vs[NNTOK][DDIM];   // 18.8 KB
    const float scale = 0.20412414523193154f;  // 1/sqrt(24)

    for (int idx = threadIdx.x; idx < NNTOK * DDIM; idx += 256) {
        int n = idx / DDIM, e = idx % DDIM;
        size_t base = ((size_t)b * NNTOK + n) * (3 * EE);
        Ks[n][e] = qkv[base + EE     + h * DDIM + e];
        Vs[n][e] = qkv[base + 2 * EE + h * DDIM + e];
    }
    __syncthreads();

    for (int q = threadIdx.x; q < NNTOK; q += 256) {
        float qv[DDIM];
        const float* qrow = qkv + ((size_t)b * NNTOK + q) * (3 * EE) + h * DDIM;
        #pragma unroll
        for (int e = 0; e < DDIM; ++e) qv[e] = qrow[e] * scale;

        float mx = -3.0e38f;
        for (int n = 0; n < NNTOK; ++n) {
            float s = 0.f;
            #pragma unroll
            for (int e = 0; e < DDIM; ++e) s += qv[e] * Ks[n][e];
            mx = fmaxf(mx, s);
        }
        float sum = 0.f;
        float acc[DDIM];
        #pragma unroll
        for (int e = 0; e < DDIM; ++e) acc[e] = 0.f;
        for (int n = 0; n < NNTOK; ++n) {
            float s = 0.f;
            #pragma unroll
            for (int e = 0; e < DDIM; ++e) s += qv[e] * Ks[n][e];
            float ev = __expf(s - mx);
            sum += ev;
            #pragma unroll
            for (int e = 0; e < DDIM; ++e) acc[e] += ev * Vs[n][e];
        }
        float inv = __builtin_amdgcn_rcpf(sum);
        _Float16* orow = o16 + ((size_t)b * NNTOK + q) * EE + h * DDIM;
        #pragma unroll
        for (int e = 0; e < DDIM; ++e) orow[e] = (_Float16)(acc[e] * inv);
    }
}

// ---------------- elementwise helpers ----------------
__global__ void add_inplace_kernel(float* __restrict__ dst, const float* __restrict__ src, size_t n)
{
    size_t i = (size_t)blockIdx.x * blockDim.x + threadIdx.x;
    if (i < n) dst[i] += src[i];
}

__global__ void add_time_f16_kernel(float* __restrict__ p, const float* __restrict__ timev,
                                    _Float16* __restrict__ out16)
{
    size_t i = (size_t)blockIdx.x * blockDim.x + threadIdx.x;
    if (i >= (size_t)NT * EE) return;
    float v = p[i] + timev[i % EE];
    p[i] = v;
    out16[i] = (_Float16)v;
}

// tokens[B,N,E] -> nchw[B,E,14,14]
__global__ void tokens_to_nchw_kernel(const float* __restrict__ p, float* __restrict__ f)
{
    size_t idx = (size_t)blockIdx.x * blockDim.x + threadIdx.x;
    if (idx >= (size_t)BB * EE * NNTOK) return;
    int rc = (int)(idx % NNTOK);
    size_t t = idx / NNTOK;
    int e = (int)(t % EE);
    int b = (int)(t / EE);
    f[idx] = p[((size_t)b * NNTOK + rc) * EE + e];
}

// ---------------- transposed conv (stride2, k4, pad1 equiv; w: [Cin,Cout,4,4]) -----
// mode 0: raw ; mode 1: silu then clip[0,1]
__global__ void convt_kernel(const float* __restrict__ x, const float* __restrict__ w,
                             const float* __restrict__ bias, float* __restrict__ y,
                             int Bn, int Cin, int Cout, int Hin, int Win, int mode)
{
    int Hout = Hin * 2, Wout = Win * 2;
    size_t total = (size_t)Bn * Cout * Hout * Wout;
    size_t idx = (size_t)blockIdx.x * blockDim.x + threadIdx.x;
    if (idx >= total) return;
    int ox = (int)(idx % Wout); size_t t = idx / Wout;
    int oy = (int)(t % Hout);   t /= Hout;
    int co = (int)(t % Cout);
    int b  = (int)(t / Cout);

    float acc = bias[co];
    for (int kh = (oy + 1) & 1; kh < 4; kh += 2) {
        int ih2 = oy + 1 - kh;
        if (ih2 < 0) continue;
        int ih = ih2 >> 1;
        if (ih >= Hin) continue;
        for (int kw = (ox + 1) & 1; kw < 4; kw += 2) {
            int iw2 = ox + 1 - kw;
            if (iw2 < 0) continue;
            int iw = iw2 >> 1;
            if (iw >= Win) continue;
            const float* xp = x + (((size_t)b * Cin) * Hin + ih) * Win + iw;
            const float* wp = w + ((size_t)co * 4 + kh) * 4 + kw;
            for (int ci = 0; ci < Cin; ++ci)
                acc += xp[(size_t)ci * Hin * Win] * wp[(size_t)ci * Cout * 16];
        }
    }
    if (mode == 1) {
        acc = silu_f(acc);
        acc = fminf(fmaxf(acc, 0.0f), 1.0f);
    }
    y[idx] = acc;
}

// ---------------- spatial LayerNorm (per b,c over HxW, per-pixel affine) + SiLU ----
__global__ __launch_bounds__(256)
void ln_spatial_silu_kernel(float* __restrict__ f, const float* __restrict__ g,
                            const float* __restrict__ bt, int HW)
{
    float* row = f + (size_t)blockIdx.x * HW;
    __shared__ float red[256];
    int tid = threadIdx.x;

    float s = 0.f;
    for (int i = tid; i < HW; i += 256) s += row[i];
    red[tid] = s; __syncthreads();
    for (int st = 128; st > 0; st >>= 1) { if (tid < st) red[tid] += red[tid + st]; __syncthreads(); }
    float mu = red[0] / (float)HW;
    __syncthreads();

    float v = 0.f;
    for (int i = tid; i < HW; i += 256) { float d = row[i] - mu; v += d * d; }
    red[tid] = v; __syncthreads();
    for (int st = 128; st > 0; st >>= 1) { if (tid < st) red[tid] += red[tid + st]; __syncthreads(); }
    float rs = rsqrtf(red[0] / (float)HW + 1e-5f);
    __syncthreads();

    for (int i = tid; i < HW; i += 256) {
        float val = (row[i] - mu) * rs * g[i] + bt[i];
        row[i] = silu_f(val);
    }
}

// ---------------- workspace layout (peak ~168.6 MB) ----------------
static constexpr size_t OFF_P     = 0;                         //  9,633,792  f32 residual stream
static constexpr size_t OFF_H16   = OFF_P    + 9633792ULL;     //  4,816,896  f16 buffer A
static constexpr size_t OFF_H16B  = OFF_H16  + 4816896ULL;     //  9,633,792 region: f16 buffer B (4.8MB used)
static constexpr size_t OFF_TMP   = OFF_H16B + 9633792ULL;     //  9,633,792  f32 proj out
static constexpr size_t OFF_QKV   = OFF_TMP  + 9633792ULL;     // 28,901,376
static constexpr size_t OFF_A16   = OFF_QKV  + 28901376ULL;    // 12,845,056  im2col f16
static constexpr size_t OFF_WBP   = OFF_A16  + 12845056ULL;    // f16 weights (stored as B^T[N][K])
static constexpr size_t OFF_WQKV0 = OFF_WBP   + 49152ULL;
static constexpr size_t OFF_WQKV1 = OFF_WQKV0 + 55296ULL;
static constexpr size_t OFF_WO0   = OFF_WQKV1 + 55296ULL;
static constexpr size_t OFF_WO1   = OFF_WO0   + 18432ULL;
static constexpr size_t OFF_WM0   = OFF_WO1   + 18432ULL;
static constexpr size_t OFF_WM1   = OFF_WM0   + 18432ULL;
static constexpr size_t OFF_WM2   = OFF_WM1   + 18432ULL;
static constexpr size_t OFF_TIME  = OFF_WM2   + 18432ULL;
// decoder overlays (encoder scratch is dead by then)
static constexpr size_t OFF_FIN   = OFF_TMP;                   //  9,633,792  [B,96,14,14]
static constexpr size_t OFF_F1    = OFF_QKV;                   // 19,267,584  [B,48,28,28]
static constexpr size_t OFF_F2    = 53000192ULL;               // 38,535,168  [B,24,56,56]
static constexpr size_t OFF_F3    = 91535360ULL;               // 77,070,336  [B,12,112,112]

static inline unsigned cdiv256(size_t n) { return (unsigned)((n + 255) / 256); }

extern "C" void kernel_launch(void* const* d_in, const int* in_sizes, int n_in,
                              void* d_out, int out_size, void* d_ws, size_t ws_size,
                              hipStream_t stream)
{
    (void)in_sizes; (void)n_in; (void)out_size; (void)ws_size;

    const float* x   = (const float*)d_in[0];
    const float* ts  = (const float*)d_in[1];
    const float* Wp  = (const float*)d_in[2];
    const float* bp  = (const float*)d_in[3];
    const float* Wt1 = (const float*)d_in[4];
    const float* bt1 = (const float*)d_in[5];
    const float* Wt2 = (const float*)d_in[6];
    const float* bt2 = (const float*)d_in[7];
    const float* a_g[2]    = {(const float*)d_in[8],  (const float*)d_in[14]};
    const float* a_b[2]    = {(const float*)d_in[9],  (const float*)d_in[15]};
    const float* a_Wqkv[2] = {(const float*)d_in[10], (const float*)d_in[16]};
    const float* a_bqkv[2] = {(const float*)d_in[11], (const float*)d_in[17]};
    const float* a_Wo[2]   = {(const float*)d_in[12], (const float*)d_in[18]};
    const float* a_bo[2]   = {(const float*)d_in[13], (const float*)d_in[19]};
    const float* Wm[3] = {(const float*)d_in[20], (const float*)d_in[22], (const float*)d_in[24]};
    const float* bm[3] = {(const float*)d_in[21], (const float*)d_in[23], (const float*)d_in[25]};
    const float* Wd1 = (const float*)d_in[26]; const float* bd1 = (const float*)d_in[27];
    const float* l1g = (const float*)d_in[28]; const float* l1b = (const float*)d_in[29];
    const float* Wd2 = (const float*)d_in[30]; const float* bd2 = (const float*)d_in[31];
    const float* l2g = (const float*)d_in[32]; const float* l2b = (const float*)d_in[33];
    const float* Wd3 = (const float*)d_in[34]; const float* bd3 = (const float*)d_in[35];
    const float* l3g = (const float*)d_in[36]; const float* l3b = (const float*)d_in[37];
    const float* Wd4 = (const float*)d_in[38]; const float* bd4 = (const float*)d_in[39];

    char* ws = (char*)d_ws;
    float*    p    = (float*)(ws + OFF_P);
    _Float16* h16a = (_Float16*)(ws + OFF_H16);
    _Float16* h16b = (_Float16*)(ws + OFF_H16B);
    float*    tmp  = (float*)(ws + OFF_TMP);
    float*    qkv  = (float*)(ws + OFF_QKV);
    _Float16* A16  = (_Float16*)(ws + OFF_A16);
    _Float16* wBP  = (_Float16*)(ws + OFF_WBP);
    _Float16* wQKV[2] = {(_Float16*)(ws + OFF_WQKV0), (_Float16*)(ws + OFF_WQKV1)};
    _Float16* wO[2]   = {(_Float16*)(ws + OFF_WO0),   (_Float16*)(ws + OFF_WO1)};
    _Float16* wM[3]   = {(_Float16*)(ws + OFF_WM0), (_Float16*)(ws + OFF_WM1), (_Float16*)(ws + OFF_WM2)};
    float*    timev = (float*)(ws + OFF_TIME);
    float*    fin = (float*)(ws + OFF_FIN);
    float*    f1  = (float*)(ws + OFF_F1);
    float*    f2  = (float*)(ws + OFF_F2);
    float*    f3  = (float*)(ws + OFF_F3);
    float*    out = (float*)d_out;

    const dim3 blk32(32);
    const dim3 gN6(NT / 64, EE / 16);          // N = 96
    const dim3 gN18(NT / 64, (3 * EE) / 16);   // N = 288

    // ---- time embedding ----
    time_embed_kernel<<<1, 128, 0, stream>>>(ts, Wt1, bt1, Wt2, bt2, timev);

    // ---- weight conversions to f16, stored as B^T[N][K] ----
    // patch GEMM: B[k][e] = Wp[e][k]  =>  B^T[e][k] = Wp[e][k]  (plain copy)
    f32_to_f16_kernel<<<cdiv256((size_t)EE * KPATCH), 256, 0, stream>>>(Wp, wBP, (size_t)EE * KPATCH);
    // Wqkv[k][n], Wo[k][n], Wm[k][n]: B^T[n][k] = W[k][n]  (transpose)
    for (int i = 0; i < 2; ++i) {
        transpose_f32_to_f16_kernel<<<cdiv256((size_t)EE * 3 * EE), 256, 0, stream>>>(a_Wqkv[i], wQKV[i], EE, 3 * EE);
        transpose_f32_to_f16_kernel<<<cdiv256((size_t)EE * EE), 256, 0, stream>>>(a_Wo[i], wO[i], EE, EE);
    }
    for (int i = 0; i < 3; ++i)
        transpose_f32_to_f16_kernel<<<cdiv256((size_t)EE * EE), 256, 0, stream>>>(Wm[i], wM[i], EE, EE);

    // ---- patch embed: im2col + WMMA GEMM [25088,256]x[256,96] + bp -> p (f32) ----
    im2col_patch_kernel<<<cdiv256((size_t)NT * KPATCH), 256, 0, stream>>>(x, A16);
    wmma_gemm_t<KPATCH, 0, false><<<gN6, blk32, 0, stream>>>(A16, wBP, bp, p, nullptr, EE);

    // ---- rotary embedding (in place) ----
    rope_kernel<<<cdiv256((size_t)NT * (EE / 2)), 256, 0, stream>>>(p);

    // ---- two MHSA blocks ----
    for (int b = 0; b < 2; ++b) {
        ln_tokens_f16_kernel<<<cdiv256(NT), 256, 0, stream>>>(p, a_g[b], a_b[b], h16a);
        wmma_gemm_t<EE, 0, false><<<gN18, blk32, 0, stream>>>(h16a, wQKV[b], a_bqkv[b], qkv, nullptr, 3 * EE);
        attention_kernel<<<BB * HHEADS, 256, 0, stream>>>(qkv, h16a);   // f16 out
        wmma_gemm_t<EE, 0, false><<<gN6, blk32, 0, stream>>>(h16a, wO[b], a_bo[b], tmp, nullptr, EE);
        add_inplace_kernel<<<cdiv256((size_t)NT * EE), 256, 0, stream>>>(p, tmp, (size_t)NT * EE);
    }

    // ---- + time (broadcast), fused with f16 emit for MLP0 ----
    add_time_f16_kernel<<<cdiv256((size_t)NT * EE), 256, 0, stream>>>(p, timev, h16a);

    // ---- MLP: three SiLU GEMMs; intermediates stay f16 (h16a <-> h16b) ----
    wmma_gemm_t<EE, 1, true ><<<gN6, blk32, 0, stream>>>(h16a, wM[0], bm[0], nullptr, h16b, EE);
    wmma_gemm_t<EE, 1, true ><<<gN6, blk32, 0, stream>>>(h16b, wM[1], bm[1], nullptr, h16a, EE);
    wmma_gemm_t<EE, 1, false><<<gN6, blk32, 0, stream>>>(h16a, wM[2], bm[2], p, nullptr, EE);

    // ---- decoder ----
    tokens_to_nchw_kernel<<<cdiv256((size_t)BB * EE * NNTOK), 256, 0, stream>>>(p, fin);

    // convT1: [B,96,14,14] -> [B,48,28,28] ; LN(28x28) + SiLU
    convt_kernel<<<cdiv256((size_t)BB * 48 * 28 * 28), 256, 0, stream>>>(fin, Wd1, bd1, f1, BB, 96, 48, 14, 14, 0);
    ln_spatial_silu_kernel<<<BB * 48, 256, 0, stream>>>(f1, l1g, l1b, 28 * 28);

    // convT2: -> [B,24,56,56] ; LN + SiLU
    convt_kernel<<<cdiv256((size_t)BB * 24 * 56 * 56), 256, 0, stream>>>(f1, Wd2, bd2, f2, BB, 48, 24, 28, 28, 0);
    ln_spatial_silu_kernel<<<BB * 24, 256, 0, stream>>>(f2, l2g, l2b, 56 * 56);

    // convT3: -> [B,12,112,112] ; LN + SiLU
    convt_kernel<<<cdiv256((size_t)BB * 12 * 112 * 112), 256, 0, stream>>>(f2, Wd3, bd3, f3, BB, 24, 12, 56, 56, 0);
    ln_spatial_silu_kernel<<<BB * 12, 256, 0, stream>>>(f3, l3g, l3b, 112 * 112);

    // convT4: -> [B,1,224,224] ; SiLU + clip fused
    convt_kernel<<<cdiv256((size_t)BB * 1 * 224 * 224), 256, 0, stream>>>(f3, Wd4, bd4, out, BB, 12, 1, 112, 112, 1);
}
